// MoeLayer_8272107012499
// MI455X (gfx1250) — matrix operand order repaired and checked by reading, not compiled, not run on hardware
//
#include <hip/hip_runtime.h>
#include <hip/hip_bf16.h>

#define NTOK 16384
#define DIN  1024
#define DEXP 2048
#define DOUT 1024
#define NE   8
#define LIST_CAP  (2*NTOK + NE*16)   /* 32896 */
#define MAX_TILES (LIST_CAP/16)      /* 2056  */
#define SMEM_BYTES 66752

typedef __attribute__((ext_vector_type(8)))  float  v8f;
typedef __attribute__((ext_vector_type(8)))  __bf16 v8bf;
typedef __attribute__((ext_vector_type(16))) __bf16 v16bf;

__device__ __forceinline__ __bf16 f2bf(float f) {
  unsigned u = __float_as_uint(f);
  unsigned r = (u + 0x7FFFu + ((u >> 16) & 1u)) >> 16;
  unsigned short s = (unsigned short)r;
  __bf16 b; __builtin_memcpy(&b, &s, 2);
  return b;
}

// ---------------- init: zero counters, list = -1 ----------------
__global__ void init_kernel(int* counts, int* cursor, int* list) {
  int i = blockIdx.x * blockDim.x + threadIdx.x;
  if (i < NE) { counts[i] = 0; cursor[i] = 0; }
  for (int j = i; j < LIST_CAP; j += gridDim.x * blockDim.x) list[j] = -1;
}

// ---------------- f32 -> bf16 convert (grid-stride) ----------------
__global__ void cvt_kernel(const float* __restrict__ in, __bf16* __restrict__ out, size_t n) {
  size_t i = (size_t)blockIdx.x * blockDim.x + threadIdx.x;
  size_t stride = (size_t)gridDim.x * blockDim.x;
  for (; i < n; i += stride) out[i] = f2bf(in[i]);
}

// ---------------- router: logits, softmax, top-2, counts ----------------
__global__ __launch_bounds__(256)
void router_kernel(const float* __restrict__ x, const float* __restrict__ Wg,
                   const float* __restrict__ bg, int* __restrict__ kidx,
                   float* __restrict__ kw, float* __restrict__ probs,
                   int* __restrict__ counts) {
  __shared__ float sWg[NE * DIN];                    // 32 KB
  for (int i = threadIdx.x; i < NE * DIN; i += blockDim.x) sWg[i] = Wg[i];
  __syncthreads();
  int t = blockIdx.x * blockDim.x + threadIdx.x;     // N is a multiple of 256
  float acc[NE];
#pragma unroll
  for (int e = 0; e < NE; ++e) acc[e] = bg[e];
  const float4* xr  = (const float4*)(x + (size_t)t * DIN);
  const float4* wg4 = (const float4*)sWg;
  for (int k = 0; k < DIN / 4; ++k) {
    float4 xv = xr[k];
#pragma unroll
    for (int e = 0; e < NE; ++e) {
      float4 wv = wg4[e * (DIN / 4) + k];
      acc[e] += xv.x * wv.x + xv.y * wv.y + xv.z * wv.z + xv.w * wv.w;
    }
  }
  float mx = acc[0];
#pragma unroll
  for (int e = 1; e < NE; ++e) mx = fmaxf(mx, acc[e]);
  float p[NE]; float s = 0.f;
#pragma unroll
  for (int e = 0; e < NE; ++e) { p[e] = __expf(acc[e] - mx); s += p[e]; }
  float inv = 1.f / s;
#pragma unroll
  for (int e = 0; e < NE; ++e) { p[e] *= inv; probs[(size_t)t * NE + e] = p[e]; }
  int i1 = 0; float v1 = p[0];
#pragma unroll
  for (int e = 1; e < NE; ++e) if (p[e] > v1) { v1 = p[e]; i1 = e; }
  int i2 = -1; float v2 = -1.f;
#pragma unroll
  for (int e = 0; e < NE; ++e) if (e != i1 && p[e] > v2) { v2 = p[e]; i2 = e; }
  float wsum = v1 + v2;
  kidx[t * 2 + 0] = i1; kidx[t * 2 + 1] = i2;
  kw[t * 2 + 0] = v1 / wsum; kw[t * 2 + 1] = v2 / wsum;
  atomicAdd(&counts[i1], 1);
  atomicAdd(&counts[i2], 1);
}

// ---------------- padded offsets (single thread) ----------------
__global__ void offsets_kernel(const int* counts, int* poff) {
  if (threadIdx.x == 0 && blockIdx.x == 0) {
    int o = 0;
    for (int e = 0; e < NE; ++e) { poff[e] = o; o += (counts[e] + 15) & ~15; }
    poff[NE] = o;
  }
}

// ---------------- bucket (token,slot) pairs per expert ----------------
__global__ void fill_kernel(const int* __restrict__ kidx, const int* __restrict__ poff,
                            int* cursor, int* list) {
  int i = blockIdx.x * blockDim.x + threadIdx.x;     // over 2N
  if (i >= 2 * NTOK) return;
  int e = kidx[i];
  int pos = atomicAdd(&cursor[e], 1);
  list[poff[e] + pos] = i;                           // i = token*2 + slot
}

// ---------------- deterministic load-balancing loss ----------------
__global__ __launch_bounds__(256)
void loss_kernel(const float* __restrict__ probs, const int* __restrict__ counts,
                 float* loss_out) {
  __shared__ float red[256 * NE];
  float s[NE];
#pragma unroll
  for (int e = 0; e < NE; ++e) s[e] = 0.f;
  for (int t = threadIdx.x; t < NTOK; t += 256)
#pragma unroll
    for (int e = 0; e < NE; ++e) s[e] += probs[(size_t)t * NE + e];
#pragma unroll
  for (int e = 0; e < NE; ++e) red[threadIdx.x * NE + e] = s[e];
  __syncthreads();
  for (int off = 128; off > 0; off >>= 1) {
    if ((int)threadIdx.x < off)
#pragma unroll
      for (int e = 0; e < NE; ++e)
        red[threadIdx.x * NE + e] += red[(threadIdx.x + off) * NE + e];
    __syncthreads();
  }
  if (threadIdx.x == 0) {
    float L = 0.f;
    for (int e = 0; e < NE; ++e)
      L += ((float)counts[e] / (float)NTOK) * (red[e] / (float)NTOK);
    loss_out[0] = (float)NE * L;
  }
}

// ---------------- fused expert megakernel: GEMM1 + SoLU + GEMM2 ----------------
__global__ __launch_bounds__(512)
void expert_kernel(const __bf16* __restrict__ x_bf,
                   const __bf16* __restrict__ win_bf,
                   const float*  __restrict__ b_in,
                   const __bf16* __restrict__ wout_bf,
                   const int* __restrict__ list,
                   const int* __restrict__ poff,
                   float* __restrict__ out_slots) {
  extern __shared__ char smem[];
  __bf16* acts  = (__bf16*)smem;                     // 16*2048*2 = 65536 B
  float* red    = (float*)(smem + 65536);            // 16 waves * 16 rows
  float* rowmax = (float*)(smem + 66560);            // 16
  float* rowsum = (float*)(smem + 66624);            // 16
  int*   rows   = (int*)  (smem + 66688);            // 16

  int base = blockIdx.x * 16;
  int e = -1;
#pragma unroll
  for (int i = 0; i < NE; ++i)
    if (base >= poff[i] && base < poff[i + 1]) e = i;
  if (e < 0) return;                                 // uniform: past padded total

  int lane = threadIdx.x & 31;
  int wave = threadIdx.x >> 5;
  int m16  = lane & 15;                              // A row / B column / D column
  int h    = lane >> 4;                              // K-half selector

  if (threadIdx.x < 16) rows[threadIdx.x] = list[base + threadIdx.x];
  __syncthreads();

  int tok_row = rows[m16];
  int tokA = (tok_row < 0) ? 0 : (tok_row >> 1);
  const __bf16* xrow = x_bf + (size_t)tokA * DIN;

  // ---- GEMM1: h16x2048 = x16x1024 . W_in^T ; wave owns 128 hidden cols ----
  v8f c[8];
  const v8f vz = {0.f,0.f,0.f,0.f,0.f,0.f,0.f,0.f};
#pragma unroll
  for (int j = 0; j < 8; ++j) c[j] = vz;
  const __bf16* Wb = win_bf + (size_t)e * DEXP * DIN;
  for (int kb = 0; kb < DIN; kb += 32) {
    v8bf a0 = *(const v8bf*)(xrow + kb + h * 16);         // K = kb+h*16 .. +7
    v8bf a1 = *(const v8bf*)(xrow + kb + h * 16 + 8);     // K = kb+h*16+8 .. +15
    v16bf a = __builtin_shufflevector(a0, a1, 0,1,2,3,4,5,6,7,8,9,10,11,12,13,14,15);
#pragma unroll
    for (int j = 0; j < 8; ++j) {
      int n = wave * 128 + j * 16 + m16;
      v16bf b = *(const v16bf*)(Wb + (size_t)n * DIN + kb + h * 16);
      c[j] = __builtin_amdgcn_wmma_f32_16x16x32_bf16(false, a, false, b,
                                                     (short)0, c[j], false, false);
    }
  }
  // bias (depends only on hidden col n)
  const float* brow = b_in + (size_t)e * DEXP;
#pragma unroll
  for (int j = 0; j < 8; ++j) {
    float bias = brow[wave * 128 + j * 16 + m16];
#pragma unroll
    for (int v = 0; v < 8; ++v) c[j][v] += bias;
  }

  // ---- SoLU: softmax over hidden (2048) * h ; rows distributed across waves ----
  // per-lane partial row max (element v -> row h*8+v), reduce across 16 lanes
  float mx[8];
#pragma unroll
  for (int v = 0; v < 8; ++v) {
    float m = c[0][v];
#pragma unroll
    for (int j = 1; j < 8; ++j) m = fmaxf(m, c[j][v]);
    for (int off = 1; off < 16; off <<= 1) m = fmaxf(m, __shfl_xor(m, off, 32));
    mx[v] = m;
  }
  if (m16 == 0)
#pragma unroll
    for (int v = 0; v < 8; ++v) red[wave * 16 + h * 8 + v] = mx[v];
  __syncthreads();
  if (threadIdx.x < 16) {
    float m = red[threadIdx.x];
    for (int w = 1; w < 16; ++w) m = fmaxf(m, red[w * 16 + threadIdx.x]);
    rowmax[threadIdx.x] = m;
  }
  __syncthreads();
#pragma unroll
  for (int v = 0; v < 8; ++v) {
    float rm = rowmax[h * 8 + v];
    float s = 0.f;
#pragma unroll
    for (int j = 0; j < 8; ++j) s += __expf(c[j][v] - rm);
    for (int off = 1; off < 16; off <<= 1) s += __shfl_xor(s, off, 32);
    mx[v] = s;
  }
  if (m16 == 0)
#pragma unroll
    for (int v = 0; v < 8; ++v) red[wave * 16 + h * 8 + v] = mx[v];
  __syncthreads();
  if (threadIdx.x < 16) {
    float s = red[threadIdx.x];
    for (int w = 1; w < 16; ++w) s += red[w * 16 + threadIdx.x];
    rowsum[threadIdx.x] = s;
  }
  __syncthreads();
  // acts = softmax(h)*h, bf16 into LDS tile [16][2048]
#pragma unroll
  for (int j = 0; j < 8; ++j) {
    int n = wave * 128 + j * 16 + m16;
#pragma unroll
    for (int v = 0; v < 8; ++v) {
      int m = h * 8 + v;
      float val = c[j][v];
      float a = __expf(val - rowmax[m]) / rowsum[m] * val;
      acts[m * DEXP + n] = f2bf(a);
    }
  }
  __syncthreads();

  // ---- GEMM2: out16x1024 = acts16x2048 . W_out^T ; wave owns 64 out cols ----
  v8f d[4];
#pragma unroll
  for (int j = 0; j < 4; ++j) d[j] = vz;
  const __bf16* W2 = wout_bf + (size_t)e * DOUT * DEXP;
  for (int kb = 0; kb < DEXP; kb += 32) {
    v8bf a0 = *(const v8bf*)(acts + m16 * DEXP + kb + h * 16);
    v8bf a1 = *(const v8bf*)(acts + m16 * DEXP + kb + h * 16 + 8);
    v16bf a = __builtin_shufflevector(a0, a1, 0,1,2,3,4,5,6,7,8,9,10,11,12,13,14,15);
#pragma unroll
    for (int j = 0; j < 4; ++j) {
      int n = wave * 64 + j * 16 + m16;
      v16bf b = *(const v16bf*)(W2 + (size_t)n * DEXP + kb + h * 16);
      d[j] = __builtin_amdgcn_wmma_f32_16x16x32_bf16(false, a, false, b,
                                                     (short)0, d[j], false, false);
    }
  }
  // scatter per-slot rows (padding rows skipped)
#pragma unroll
  for (int j = 0; j < 4; ++j) {
    int n = wave * 64 + j * 16 + m16;
#pragma unroll
    for (int v = 0; v < 8; ++v) {
      int m = h * 8 + v;
      int r = rows[m];
      if (r >= 0) out_slots[(size_t)r * DOUT + n] = d[j][v];
    }
  }
}

// ---------------- combine: out[t] = w0*slot0 + w1*slot1 ----------------
__global__ void combine_kernel(const float* __restrict__ out_slots,
                               const float* __restrict__ kw,
                               float* __restrict__ out) {
  unsigned i = blockIdx.x * blockDim.x + threadIdx.x;   // over N*DOUT
  unsigned t = i >> 10, o = i & 1023;
  out[i] = kw[t * 2 + 0] * out_slots[(size_t)(t * 2 + 0) * DOUT + o] +
           kw[t * 2 + 1] * out_slots[(size_t)(t * 2 + 1) * DOUT + o];
}

extern "C" void kernel_launch(void* const* d_in, const int* in_sizes, int n_in,
                              void* d_out, int out_size, void* d_ws, size_t ws_size,
                              hipStream_t stream) {
  const float* x    = (const float*)d_in[0];
  const float* Wg   = (const float*)d_in[1];
  const float* bg   = (const float*)d_in[2];
  const float* Win  = (const float*)d_in[3];
  const float* bin  = (const float*)d_in[4];
  const float* Wout = (const float*)d_in[5];
  float* out = (float*)d_out;

  char* ws = (char*)d_ws;
  __bf16* x_bf      = (__bf16*)(ws + 0);           // 33554432 B
  __bf16* win_bf    = (__bf16*)(ws + 33554432);    // 33554432 B
  __bf16* wout_bf   = (__bf16*)(ws + 67108864);    // 33554432 B
  float*  out_slots = (float*) (ws + 100663296);   // 134217728 B
  float*  probs     = (float*) (ws + 234881024);   // 524288 B
  int*    kidx      = (int*)   (ws + 235405312);   // 131072 B
  float*  kw        = (float*) (ws + 235536384);   // 131072 B
  int*    list      = (int*)   (ws + 235667456);   // 131584 B
  int*    counts    = (int*)   (ws + 235799040);
  int*    cursor    = (int*)   (ws + 235799296);
  int*    poff      = (int*)   (ws + 235799552);

  init_kernel<<<129, 256, 0, stream>>>(counts, cursor, list);
  cvt_kernel<<<2048, 256, 0, stream>>>(x,    x_bf,    (size_t)NTOK * DIN);
  cvt_kernel<<<2048, 256, 0, stream>>>(Win,  win_bf,  (size_t)NE * DEXP * DIN);
  cvt_kernel<<<2048, 256, 0, stream>>>(Wout, wout_bf, (size_t)NE * DOUT * DEXP);
  router_kernel<<<NTOK / 256, 256, 0, stream>>>(x, Wg, bg, kidx, kw, probs, counts);
  offsets_kernel<<<1, 1, 0, stream>>>(counts, poff);
  fill_kernel<<<(2 * NTOK) / 256, 256, 0, stream>>>(kidx, poff, cursor, list);
  loss_kernel<<<1, 256, 0, stream>>>(probs, counts, out + (size_t)NTOK * DOUT);
  (void)hipFuncSetAttribute((const void*)expert_kernel,
                            hipFuncAttributeMaxDynamicSharedMemorySize, SMEM_BYTES);
  expert_kernel<<<MAX_TILES, 512, SMEM_BYTES, stream>>>(x_bf, win_bf, bin, wout_bf,
                                                        list, poff, out_slots);
  combine_kernel<<<(NTOK * DOUT) / 256, 256, 0, stream>>>(out_slots, kw, out);
}
